// PointsToBEV_29429115912632
// MI455X (gfx1250) — compile-verified, matrix operand order checked by reading, not run on hardware
//
#include <hip/hip_runtime.h>
#include <hip/hip_bf16.h>

typedef __attribute__((ext_vector_type(16))) _Float16 v16h;
typedef __attribute__((ext_vector_type(8)))  float    v8f;

#define BEV_H   128
#define BEV_W   128
#define HW      16384      // 128*128
#define PTS_C   80
#define KPAD    96         // 80 padded to 3x32 for f16 WMMA K-steps
#define BEV_C   128
#define NB      4
#define NP      200000
#define NROWS_P (NB * NP)          // 800000 point rows
#define PTILES  (NROWS_P / 16)     // 50000
#define NROWS_B (NB * HW)          // 65536 bev rows
#define MTILES  (NROWS_B / 16)     // 4096
#define BN_EPS  1e-5f

// K index inside a 16x32 f16 A/B fragment for (lane>>4, element e of v16h)
// per CDNA5 ISA 7.12.2: VGPR v (=e/2) pair p (=e&1):
//   K = (v<4 ? 0 : 16) + half*8 + 2*(v&3) + p
__device__ __forceinline__ int frag_k(int half, int e) {
    int v = e >> 1, p = e & 1;
    return ((v < 4) ? 0 : 16) + (half << 3) + ((v & 3) << 1) + p;
}

// ---------------------------------------------------------------------------
// Prep: transpose/pad weights to f16 [out][KPAD] for contiguous fragment loads
// ---------------------------------------------------------------------------
__global__ void prep_weights(const float* __restrict__ w2,      // (80,80) [k][o]
                             const float* __restrict__ conv_w,  // (128,80) [o][k]
                             _Float16* __restrict__ w2h,        // [80][96]
                             _Float16* __restrict__ cwh) {      // [128][96]
    int idx = blockIdx.x * blockDim.x + threadIdx.x;
    const int n1 = PTS_C * KPAD;
    if (idx < n1) {
        int o = idx / KPAD, k = idx - o * KPAD;
        w2h[idx] = (_Float16)((k < PTS_C) ? w2[k * PTS_C + o] : 0.0f);
    } else if (idx < n1 + BEV_C * KPAD) {
        int j = idx - n1;
        int o = j / KPAD, k = j - o * KPAD;
        cwh[j] = (_Float16)((k < PTS_C) ? conv_w[o * PTS_C + k] : 0.0f);
    }
}

// ---------------------------------------------------------------------------
// Kernel 1: per-point MLP (layer1 VALU, layer2 WMMA f16) + scatter-add
// One wave handles 16 points. 5 output-channel tiles x 3 K-steps = 15 WMMAs.
// ---------------------------------------------------------------------------
__global__ __launch_bounds__(256)
void mlp_scatter(const float* __restrict__ pts,   // (B*Np, 4)
                 const float* __restrict__ w1,    // (4, 80)
                 const float* __restrict__ b1,    // (80)
                 const _Float16* __restrict__ w2h,// [80][96]
                 const float* __restrict__ b2,    // (80)
                 float* __restrict__ sums,        // (B*HW, 80)
                 float* __restrict__ cnts) {      // (B*HW)
    int wave = threadIdx.x >> 5;
    int lane = threadIdx.x & 31;
    int tile = blockIdx.x * 8 + wave;
    if (tile >= PTILES) return;
    int half = lane >> 4;
    int m    = lane & 15;
    int row  = tile * 16 + m;

    // point features (lanes 16-31 duplicate rows 0-15's loads -> L0 hit)
    float p0 = pts[row * 4 + 0];
    float p1 = pts[row * 4 + 1];
    float p2 = pts[row * 4 + 2];
    float p3 = pts[row * 4 + 3];

    // BEV cell index (mx = my = 100/128 = 0.78125 exactly)
    int ix = (int)floorf((p0 + 50.0f) / 0.78125f);
    int iy = (int)floorf((p1 + 50.0f) / 0.78125f);
    bool valid = (ix >= 0) & (ix < BEV_W) & (iy >= 0) & (iy < BEV_H);
    int ind = iy * BEV_W + ix;
    ind = min(max(ind, 0), HW - 1);
    int b  = row / NP;
    int gi = valid ? (b * HW + ind) : -1;

    if (lane < 16 && gi >= 0) atomicAdd(&cnts[gi], 1.0f);

    // broadcast cell index for the 8 C/D rows each lane will own
    int gi8[8];
#pragma unroll
    for (int r = 0; r < 8; ++r) gi8[r] = __shfl(gi, r + 8 * half, 32);

    // layer1: hidden = relu(p @ w1 + b1), written straight into A fragments
    v16h A[3];
#pragma unroll
    for (int f = 0; f < 3; ++f) {
#pragma unroll
        for (int e = 0; e < 16; ++e) {
            int k = f * 32 + frag_k(half, e);
            float h = 0.0f;
            if (k < PTS_C) {
                h = fmaf(p0, w1[k],
                    fmaf(p1, w1[PTS_C + k],
                    fmaf(p2, w1[2 * PTS_C + k],
                    fmaf(p3, w1[3 * PTS_C + k], b1[k]))));
                h = fmaxf(h, 0.0f);
            }
            A[f][e] = (_Float16)h;
        }
    }

    // layer2: 5 tiles of 16 output channels, K = 96 (3 WMMA steps)
    for (int nt = 0; nt < 5; ++nt) {
        int o = nt * 16 + m;   // N = lane&15 for B and C/D fragments
        v16h Bf[3];
#pragma unroll
        for (int f = 0; f < 3; ++f)
#pragma unroll
            for (int e = 0; e < 16; ++e)
                Bf[f][e] = w2h[o * KPAD + f * 32 + frag_k(half, e)];

        v8f acc = {};
        acc = __builtin_amdgcn_wmma_f32_16x16x32_f16(false, A[0], false, Bf[0],
                                                     (short)0, acc, false, false);
        acc = __builtin_amdgcn_wmma_f32_16x16x32_f16(false, A[1], false, Bf[1],
                                                     (short)0, acc, false, false);
        acc = __builtin_amdgcn_wmma_f32_16x16x32_f16(false, A[2], false, Bf[2],
                                                     (short)0, acc, false, false);
        float bias = b2[o];
#pragma unroll
        for (int r = 0; r < 8; ++r) {
            float v = fmaxf(acc[r] + bias, 0.0f);   // hoisted: branch body = atomic only
            int g = gi8[r];
            if (g >= 0) atomicAdd(&sums[g * PTS_C + o], v);
        }
    }
}

// ---------------------------------------------------------------------------
// Kernel 2: scatter-mean finalize -> f16 BEV features, K-padded to 96
// ---------------------------------------------------------------------------
__global__ void mean_to_f16(const float* __restrict__ sums,
                            const float* __restrict__ cnts,
                            _Float16* __restrict__ bev) {
    int idx = blockIdx.x * blockDim.x + threadIdx.x;
    if (idx >= NROWS_B * KPAD) return;
    int row = idx / KPAD, c = idx - row * KPAD;
    float v = 0.0f;
    if (c < PTS_C) v = sums[row * PTS_C + c] / fmaxf(cnts[row], 1.0f);
    bev[idx] = (_Float16)v;
}

// ---------------------------------------------------------------------------
// Kernel 3: 1x1 conv as WMMA GEMM (65536 x 128 x 80) fused with BN stats.
// block = 8 waves; wave w owns channels [16w,16w+16) of one 16-row M tile.
// Output tile is transposed through LDS so stores are 64B-coalesced in hw.
// ---------------------------------------------------------------------------
__global__ __launch_bounds__(256)
void conv_bn_stats(const _Float16* __restrict__ bev,  // [rows][96]
                   const _Float16* __restrict__ cwh,  // [128][96]
                   const float* __restrict__ conv_b,
                   float* __restrict__ out,           // (B, 128, HW) pre-BN
                   float* __restrict__ csum,
                   float* __restrict__ csq) {
    __shared__ float tr[8][16][17];   // [wave][row_local][o_local], pad 17 (coprime w/ 64 banks)

    int wave = threadIdx.x >> 5;
    int lane = threadIdx.x & 31;
    int half = lane >> 4;
    int nlo  = lane & 15;
    int mtile = blockIdx.x;              // 0..4095
    int o = wave * 16 + nlo;             // output channel, fixed per lane
    int arow = mtile * 16 + nlo;         // A-row this lane loads (M = lane&15)

    v16h Af[3], Bf[3];
#pragma unroll
    for (int f = 0; f < 3; ++f) {
#pragma unroll
        for (int e = 0; e < 16; ++e) {
            int k = f * 32 + frag_k(half, e);
            Af[f][e] = bev[arow * KPAD + k];
            Bf[f][e] = cwh[o * KPAD + k];
        }
    }
    v8f acc = {};
    acc = __builtin_amdgcn_wmma_f32_16x16x32_f16(false, Af[0], false, Bf[0],
                                                 (short)0, acc, false, false);
    acc = __builtin_amdgcn_wmma_f32_16x16x32_f16(false, Af[1], false, Bf[1],
                                                 (short)0, acc, false, false);
    acc = __builtin_amdgcn_wmma_f32_16x16x32_f16(false, Af[2], false, Bf[2],
                                                 (short)0, acc, false, false);

    float bias = conv_b[o];
    float s = 0.0f, s2 = 0.0f;
#pragma unroll
    for (int r = 0; r < 8; ++r) {
        float v = acc[r] + bias;
        s  += v;
        s2 += fmaf(v, v, 0.0f);
        tr[wave][r + 8 * half][nlo] = v;   // stage tile for transposed store
    }
    // fold the two half-wave partials (same channel o in lane and lane^16)
    s  += __shfl_xor(s, 16, 32);
    s2 += __shfl_xor(s2, 16, 32);
    if (half == 0) {
        atomicAdd(&csum[o], s);
        atomicAdd(&csq[o], s2);
    }

    __syncthreads();

    // transposed store: lanes 0-15 cover 16 consecutive hw for one channel
    int b      = (mtile * 16) >> 14;          // whole 16-row tile sits in one batch
    int hwbase = (mtile * 16) & (HW - 1);
#pragma unroll
    for (int j = 0; j < 8; ++j) {
        int o2 = wave * 16 + j + 8 * half;
        float v = tr[wave][nlo][j + 8 * half];    // stride-17 column read: bank-conflict-free
        out[((b * BEV_C + o2) << 14) + hwbase + nlo] = v;
    }
}

// ---------------------------------------------------------------------------
// Kernel 4: per-channel BN scale/shift from accumulated moments
// ---------------------------------------------------------------------------
__global__ void bn_stats(const float* __restrict__ csum,
                         const float* __restrict__ csq,
                         const float* __restrict__ gamma,
                         const float* __restrict__ beta,
                         float* __restrict__ scale,
                         float* __restrict__ shift) {
    int o = threadIdx.x;
    if (o < BEV_C) {
        const float n = (float)NROWS_B;
        float mu  = csum[o] / n;
        float var = csq[o] / n - mu * mu;
        float sc  = rsqrtf(var + BN_EPS) * gamma[o];
        scale[o] = sc;
        shift[o] = fmaf(-mu, sc, beta[o]);
    }
}

// ---------------------------------------------------------------------------
// Kernel 5: BN apply + ReLU, in place on d_out, float4 vectorized
// ---------------------------------------------------------------------------
__global__ __launch_bounds__(256)
void bn_apply(float* __restrict__ out,
              const float* __restrict__ scale,
              const float* __restrict__ shift) {
    int idx = blockIdx.x * blockDim.x + threadIdx.x;  // float4 index
    int o = (idx >> 12) & (BEV_C - 1);                // 4096 float4 per plane
    float sc = scale[o], sh = shift[o];
    float4* p = (float4*)out;
    float4 v = p[idx];
    v.x = fmaxf(fmaf(v.x, sc, sh), 0.0f);
    v.y = fmaxf(fmaf(v.y, sc, sh), 0.0f);
    v.z = fmaxf(fmaf(v.z, sc, sh), 0.0f);
    v.w = fmaxf(fmaf(v.w, sc, sh), 0.0f);
    p[idx] = v;
}

// ---------------------------------------------------------------------------
extern "C" void kernel_launch(void* const* d_in, const int* in_sizes, int n_in,
                              void* d_out, int out_size, void* d_ws, size_t ws_size,
                              hipStream_t stream) {
    const float* points = (const float*)d_in[0];
    const float* w1     = (const float*)d_in[1];
    const float* b1     = (const float*)d_in[2];
    const float* w2     = (const float*)d_in[3];
    const float* b2     = (const float*)d_in[4];
    const float* conv_w = (const float*)d_in[5];
    const float* conv_b = (const float*)d_in[6];
    const float* gamma  = (const float*)d_in[7];
    const float* beta   = (const float*)d_in[8];
    (void)in_sizes; (void)n_in; (void)out_size; (void)ws_size;

    char* ws = (char*)d_ws;
    size_t off = 0;
    float*    sums = (float*)(ws + off);    off += (size_t)NROWS_B * PTS_C * 4;   // 20.97 MB
    float*    cnts = (float*)(ws + off);    off += (size_t)NROWS_B * 4;           // 256 KB
    size_t zero_bytes = off;                                                      // sums+cnts contiguous
    _Float16* bev  = (_Float16*)(ws + off); off += (size_t)NROWS_B * KPAD * 2;    // 12.58 MB
    _Float16* w2h  = (_Float16*)(ws + off); off += (size_t)PTS_C * KPAD * 2;
    _Float16* cwh  = (_Float16*)(ws + off); off += (size_t)BEV_C * KPAD * 2;
    float*    csum = (float*)(ws + off);    off += BEV_C * 4;
    float*    csq  = (float*)(ws + off);    off += BEV_C * 4;
    float*    scl  = (float*)(ws + off);    off += BEV_C * 4;
    float*    shf  = (float*)(ws + off);    off += BEV_C * 4;

    float* out = (float*)d_out;

    // zero accumulators (graph-capture safe)
    hipMemsetAsync(sums, 0, zero_bytes, stream);
    hipMemsetAsync(csum, 0, 2 * BEV_C * sizeof(float), stream);

    // weight prep
    {
        int total = PTS_C * KPAD + BEV_C * KPAD;
        prep_weights<<<(total + 255) / 256, 256, 0, stream>>>(w2, conv_w, w2h, cwh);
    }
    // point MLP + scatter (50000 wave-tiles, 8 waves/block)
    mlp_scatter<<<PTILES / 8, 256, 0, stream>>>(points, w1, b1, w2h, b2, sums, cnts);
    // scatter-mean -> f16
    {
        int total = NROWS_B * KPAD;
        mean_to_f16<<<(total + 255) / 256, 256, 0, stream>>>(sums, cnts, bev);
    }
    // 1x1 conv GEMM + BN moment accumulation
    conv_bn_stats<<<MTILES, 256, 0, stream>>>(bev, cwh, conv_b, out, csum, csq);
    // BN scale/shift
    bn_stats<<<1, 128, 0, stream>>>(csum, csq, gamma, beta, scl, shf);
    // BN apply + ReLU (2,097,152 float4)
    bn_apply<<<(NB * BEV_C * HW / 4) / 256, 256, 0, stream>>>(out, scl, shf);
}
